// StudentModel_12197707120733
// MI455X (gfx1250) — compile-verified, matrix operand order checked by reading, not run on hardware
//
#include <hip/hip_runtime.h>

// ---------------------------------------------------------------------------
// Dialogue-graph transformer for MI455X (gfx1250, wave32, WMMA).
// Heavy GEMMs in bf16 V_WMMA_F32_16X16X32_BF16 (f32 accumulate); softmax /
// LayerNorm / GRU gating in fp32. Pairwise GRU refactored so only 256-wide
// hx GEMMs are pairwise; node terms are per-node projections.
// GEMM waves compute 16x64 tiles; all fragment loads of a k-step are issued
// before the first WMMA so waits stagger instead of draining LOADcnt to 0.
// ---------------------------------------------------------------------------

#define DEVI __device__ __forceinline__

namespace {
constexpr int  Bq = 4, Nn = 128, Hh = 768, Pp = 256, HEADS = 12, DK = 64;
constexpr int  LAYERS = 3, NC = 17;
constexpr long ROWS = (long)Bq * Nn * Nn;        // 65536 pair rows
constexpr long NODES = (long)Bq * Nn;            // 512 node rows
}

typedef __attribute__((ext_vector_type(16))) __bf16 v16bf;
typedef __attribute__((ext_vector_type(8)))  float  v8f;

union Frag { v16bf v; unsigned short u[16]; unsigned int d[8]; };

DEVI unsigned short f2bf(float f) {             // round-to-nearest-even
  unsigned int u = __float_as_uint(f);
  u += 0x7FFFu + ((u >> 16) & 1u);
  return (unsigned short)(u >> 16);
}
// K-index mapping for 16-bit A/B fragments (ISA 7.12.2, 16-bit A 16x32):
// lanes 0-15 hold K-block base 0, lanes 16-31 base +8; VGPR v<4 -> K=2v,
// v>=4 -> K=16+2(v-4); each VGPR packs (K, K+1). Pairs of dwords are
// contiguous, so the compiler lowers each fragment to 2x global_load_b128.
DEVI int kmap(int v, int half) { return (v < 4 ? 2 * v : 16 + 2 * (v - 4)) + 8 * half; }
DEVI float sigm(float x) { return 1.0f / (1.0f + __expf(-x)); }

// ------------------------------ utility kernels ----------------------------

__global__ void k_zero(float* p, long n) {
  long i = (long)blockIdx.x * blockDim.x + threadIdx.x;
  long st = (long)gridDim.x * blockDim.x;
  for (; i < n; i += st) p[i] = 0.0f;
}

__global__ void k_cvt_bf(const float* __restrict__ s, unsigned short* __restrict__ d, long n) {
  long i = (long)blockIdx.x * blockDim.x + threadIdx.x;
  long st = (long)gridDim.x * blockDim.x;
  for (; i < n; i += st) d[i] = f2bf(s[i]);
}

// Bt[n*K + k] = W[(row0+k)*ldw + n]   (store GEMM-B transposed, bf16)
__global__ void k_prep_bt(const float* __restrict__ W, unsigned short* __restrict__ Bt,
                          int K, int Ncols, int row0, int ldw) {
  long tot = (long)K * Ncols;
  long i = (long)blockIdx.x * blockDim.x + threadIdx.x;
  long st = (long)gridDim.x * blockDim.x;
  for (; i < tot; i += st) {
    int k = (int)(i % K), n = (int)(i / K);
    Bt[i] = f2bf(W[(long)(row0 + k) * ldw + n]);
  }
}

// const_path[b,i,j,:] = [spk_emb[speakers], trn_emb[turns], pos_emb[clip(j-i)]]
__global__ void k_const_path(const int* __restrict__ spk, const int* __restrict__ trn,
                             const float* __restrict__ spk_emb, const float* __restrict__ trn_emb,
                             const float* __restrict__ pos_emb, float* __restrict__ cpath) {
  long tot = ROWS * Pp;
  long i = (long)blockIdx.x * blockDim.x + threadIdx.x;
  long st = (long)gridDim.x * blockDim.x;
  for (; i < tot; i += st) {
    int p = (int)(i & (Pp - 1));
    long r = i >> 8;
    int j = (int)(r % Nn);
    int ii = (int)((r / Nn) % Nn);
    long pr = r;                                  // flat (b,i,j)
    float v;
    if (p < 64)       v = spk_emb[spk[pr] * 64 + p];
    else if (p < 128) v = trn_emb[trn[pr] * 64 + (p - 64)];
    else {
      int d = j - ii; d = d < -5 ? -5 : (d > 5 ? 5 : d); d += 5;
      v = pos_emb[d * 128 + (p - 128)];
    }
    cpath[i] = v;
  }
}

// pn = LN(const_path + hx) over P=256, output bf16.  One wave per row.
__global__ void k_ln_paths(const float* __restrict__ cp, const float* __restrict__ hx,
                           const float* __restrict__ g, const float* __restrict__ be,
                           unsigned short* __restrict__ pn) {
  int wid = threadIdx.x >> 5, lane = threadIdx.x & 31;
  long row = (long)blockIdx.x * 8 + wid;
  const float* a = cp + row * Pp;
  const float* h = hx + row * Pp;
  float x[8], s = 0.f;
#pragma unroll
  for (int t = 0; t < 8; ++t) { x[t] = a[lane + 32 * t] + h[lane + 32 * t]; s += x[t]; }
  for (int o = 16; o; o >>= 1) s += __shfl_xor(s, o);
  float mean = s * (1.0f / Pp), vs = 0.f;
#pragma unroll
  for (int t = 0; t < 8; ++t) { float d = x[t] - mean; vs += d * d; }
  for (int o = 16; o; o >>= 1) vs += __shfl_xor(vs, o);
  float inv = rsqrtf(vs * (1.0f / Pp) + 1e-5f);
#pragma unroll
  for (int t = 0; t < 8; ++t) {
    int c = lane + 32 * t;
    pn[row * Pp + c] = f2bf((x[t] - mean) * inv * g[c] + be[c]);
  }
}

// ------------------------------ generic WMMA GEMM --------------------------
// C[M,N] = act( A[M,K](bf16) * Bt[N,K](bf16)^T + bias ).
// One wave per 16x64 output block: A fragment + 4 B fragments all issued
// before the first WMMA of each k-step (staggered load waits, back-to-back
// WMMA issue); A reused across 4 column tiles (4 f32 accumulators).
__global__ __launch_bounds__(128) void k_gemm(
    const unsigned short* __restrict__ A, const unsigned short* __restrict__ Bt,
    const float* __restrict__ bias, float* __restrict__ C,
    int M, int K, int Ncols, int act) {
  int wid = threadIdx.x >> 5, lane = threadIdx.x & 31;
  int row0 = (blockIdx.x * 4 + wid) * 16;
  if (row0 >= M) return;
  int colg = blockIdx.y * 64;
  int half = lane >> 4, ln = lane & 15;
  const unsigned short* Ap = A + (long)(row0 + ln) * K;
  const unsigned short* Bp = Bt + (long)(colg + ln) * K;
  long bstep = (long)16 * K;
  v8f acc[4] = {v8f{}, v8f{}, v8f{}, v8f{}};
  for (int k0 = 0; k0 < K; k0 += 32) {
    Frag a, b0, b1, b2, b3;
#pragma unroll
    for (int v = 0; v < 8; ++v) {
      int kb = k0 + kmap(v, half);
      a.d[v]  = *(const unsigned int*)(Ap + kb);
      b0.d[v] = *(const unsigned int*)(Bp + kb);
      b1.d[v] = *(const unsigned int*)(Bp + bstep + kb);
      b2.d[v] = *(const unsigned int*)(Bp + 2 * bstep + kb);
      b3.d[v] = *(const unsigned int*)(Bp + 3 * bstep + kb);
    }
    acc[0] = __builtin_amdgcn_wmma_f32_16x16x32_bf16(false, a.v, false, b0.v, (short)0, acc[0], false, false);
    acc[1] = __builtin_amdgcn_wmma_f32_16x16x32_bf16(false, a.v, false, b1.v, (short)0, acc[1], false, false);
    acc[2] = __builtin_amdgcn_wmma_f32_16x16x32_bf16(false, a.v, false, b2.v, (short)0, acc[2], false, false);
    acc[3] = __builtin_amdgcn_wmma_f32_16x16x32_bf16(false, a.v, false, b3.v, (short)0, acc[3], false, false);
  }
#pragma unroll
  for (int t = 0; t < 4; ++t) {
    int col = colg + t * 16 + ln;
    float bb = bias ? bias[col] : 0.0f;
#pragma unroll
    for (int g = 0; g < 8; ++g) {
      long r = row0 + g + 8 * half;               // D layout: VGPR g -> M=g(+8)
      float x = acc[t][g] + bb;
      if (act == 1) x = tanhf(x);
      C[r * Ncols + col] = x;
    }
  }
}

// ------------------------------ attention (VALU, tiny) ---------------------
// skv layout: row ((b*N+q)*N + k) * 128 : [0:64]=sk, [64:128]=sv.
__global__ __launch_bounds__(128) void k_attn(
    const float* __restrict__ q, const float* __restrict__ k, const float* __restrict__ v,
    const float* __restrict__ skv, const float* __restrict__ bias,
    float* __restrict__ out) {
  __shared__ float qs[4][64];
  __shared__ float ws[4][128];
  int wid = threadIdx.x >> 5, lane = threadIdx.x & 31;
  int flat = blockIdx.x * 4 + wid;                  // (b, h, q)
  int b = flat / (HEADS * Nn), rem = flat % (HEADS * Nn);
  int h = rem / Nn, qi = rem % Nn;
  long qrow = ((long)(b * Nn) + qi) * Hh + h * DK;
  for (int d = lane; d < DK; d += 32) qs[wid][d] = q[qrow + d] * 0.125f;  // DK^-0.5
  __syncthreads();
  const float* skbase = skv + ((long)(b * Nn + qi)) * Nn * 128;
  float lg[4];
#pragma unroll
  for (int m = 0; m < 4; ++m) {
    int kk = lane + m * 32;
    const float* kp  = k + ((long)(b * Nn) + kk) * Hh + h * DK;
    const float* skp = skbase + (long)kk * 128;
    float s = bias[b * Nn + kk];
    for (int d = 0; d < DK; ++d) s += qs[wid][d] * (kp[d] + skp[d]);
    lg[m] = s;
  }
  float mx = fmaxf(fmaxf(lg[0], lg[1]), fmaxf(lg[2], lg[3]));
  for (int o = 16; o; o >>= 1) mx = fmaxf(mx, __shfl_xor(mx, o));
  float se = 0.f;
#pragma unroll
  for (int m = 0; m < 4; ++m) { lg[m] = __expf(lg[m] - mx); se += lg[m]; }
  for (int o = 16; o; o >>= 1) se += __shfl_xor(se, o);
  float inv = 1.0f / se;
#pragma unroll
  for (int m = 0; m < 4; ++m) ws[wid][lane + m * 32] = lg[m] * inv;
  __syncthreads();
  int d0 = lane * 2;
  const float* vb  = v + (long)(b * Nn) * Hh + h * DK;
  const float* svb = skbase + 64;                 // sv half of skv row
  float o0 = 0.f, o1 = 0.f;
  for (int kk = 0; kk < Nn; ++kk) {
    float w = ws[wid][kk];
    o0 += w * (vb[(long)kk * Hh + d0]     + svb[(long)kk * 128 + d0]);
    o1 += w * (vb[(long)kk * Hh + d0 + 1] + svb[(long)kk * 128 + d0 + 1]);
  }
  out[qrow + d0] = o0;
  out[qrow + d0 + 1] = o1;
}

// nodes = LN(nodes + relu(proj)) over H=768 (proj already has bo added).
__global__ void k_postproj(const float* __restrict__ proj, float* __restrict__ nodes,
                           unsigned short* __restrict__ nodes_bf,
                           const float* __restrict__ g, const float* __restrict__ be) {
  int wid = threadIdx.x >> 5, lane = threadIdx.x & 31;
  long row = (long)blockIdx.x * 8 + wid;
  const float* pr = proj + row * Hh;
  float* np = nodes + row * Hh;
  float x[24], s = 0.f;
#pragma unroll
  for (int t = 0; t < 24; ++t) {
    float o = pr[lane + 32 * t];
    o = o > 0.f ? o : 0.f;
    x[t] = np[lane + 32 * t] + o;
    s += x[t];
  }
  for (int o = 16; o; o >>= 1) s += __shfl_xor(s, o);
  float mean = s * (1.0f / Hh), vs = 0.f;
#pragma unroll
  for (int t = 0; t < 24; ++t) { float d = x[t] - mean; vs += d * d; }
  for (int o = 16; o; o >>= 1) vs += __shfl_xor(vs, o);
  float inv = rsqrtf(vs * (1.0f / Hh) + 1e-5f);
#pragma unroll
  for (int t = 0; t < 24; ++t) {
    int c = lane + 32 * t;
    float y = (x[t] - mean) * inv * g[c] + be[c];
    np[c] = y;
    nodes_bf[row * Hh + c] = f2bf(y);
  }
}

// --------------------- fused GRU path update (WMMA) ------------------------
// One wave per 16 pair-rows (fixed b,i; j-tile). A-fragments of hx (16x256
// bf16) held in registers and reused across 3 matrices x 16 column tiles.
__global__ __launch_bounds__(128) void k_path_update(
    float* hx,
    const unsigned short* __restrict__ WrCT, const unsigned short* __restrict__ WzCT,
    const unsigned short* __restrict__ WuT,
    const float* __restrict__ NAr, const float* __restrict__ NBr,
    const float* __restrict__ NAz, const float* __restrict__ NBz,
    const float* __restrict__ NAc, const float* __restrict__ NBc,
    const float* __restrict__ br, const float* __restrict__ bz,
    const float* __restrict__ bc, const float* __restrict__ bu) {
  int wid = threadIdx.x >> 5, lane = threadIdx.x & 31;
  long tile = (long)blockIdx.x * 4 + wid;
  long base = tile * 16;
  int b = (int)(base / ((long)Nn * Nn));
  long rem = base % ((long)Nn * Nn);
  int i = (int)(rem / Nn), j0 = (int)(rem % Nn);
  int half = lane >> 4, ln = lane & 15;

  // A fragments: rows j0..j0+15 of hx (256-wide), fp32 -> bf16 in-register.
  const float* hrow = hx + (((long)(b * Nn + i)) * Nn + (j0 + ln)) * Pp;
  Frag afr[8];
#pragma unroll
  for (int s = 0; s < 8; ++s)
#pragma unroll
    for (int v = 0; v < 8; ++v) {
      int kb = s * 32 + kmap(v, half);
      afr[s].u[2 * v]     = f2bf(hrow[kb]);
      afr[s].u[2 * v + 1] = f2bf(hrow[kb + 1]);
    }

  long nbase = ((long)(b * Nn) + i) * Pp;              // NB*: node i
  for (int ct = 0; ct < 16; ++ct) {
    int c = ct * 16 + ln;
    const unsigned short* wr = WrCT + (long)c * Pp;
    const unsigned short* wz = WzCT + (long)c * Pp;
    const unsigned short* wu = WuT  + (long)c * Pp;
    v8f ar = {}, az = {}, au = {};
#pragma unroll
    for (int s = 0; s < 8; ++s) {
      Frag fr, fz, fu;
#pragma unroll
      for (int v = 0; v < 8; ++v) {
        int kb = s * 32 + kmap(v, half);
        fr.d[v] = *(const unsigned int*)(wr + kb);
        fz.d[v] = *(const unsigned int*)(wz + kb);
        fu.d[v] = *(const unsigned int*)(wu + kb);
      }
      ar = __builtin_amdgcn_wmma_f32_16x16x32_bf16(false, afr[s].v, false, fr.v, (short)0, ar, false, false);
      az = __builtin_amdgcn_wmma_f32_16x16x32_bf16(false, afr[s].v, false, fz.v, (short)0, az, false, false);
      au = __builtin_amdgcn_wmma_f32_16x16x32_bf16(false, afr[s].v, false, fu.v, (short)0, au, false, false);
    }
    float brc = br[c], bzc = bz[c], bcc = bc[c], buc = bu[c];
    float nbr = NBr[nbase + c], nbz = NBz[nbase + c], nbc = NBc[nbase + c];
#pragma unroll
    for (int g = 0; g < 8; ++g) {
      int j = j0 + g + 8 * half;                       // D layout row
      long na = ((long)(b * Nn) + j) * Pp + c;
      long pr = (((long)(b * Nn + i)) * Nn + j) * Pp + c;
      float r = sigm(NAr[na] + nbr + ar[g] + brc);
      float z = sigm(NAz[na] + nbz + az[g] + bzc);
      float u = tanhf(NAc[na] + nbc + bcc + r * (au[g] + buc));
      float h = hx[pr];
      hx[pr] = z * h + (1.0f - z) * u;
    }
  }
}

// pred_bf[row, 0:256]=const_path, [256:512]=hx  (bf16)
__global__ void k_concat_pred(const float* __restrict__ cp, const float* __restrict__ hx,
                              unsigned short* __restrict__ pred) {
  long tot = ROWS * 512;
  long i = (long)blockIdx.x * blockDim.x + threadIdx.x;
  long st = (long)gridDim.x * blockDim.x;
  for (; i < tot; i += st) {
    long row = i >> 9; int c = (int)(i & 511);
    float v = c < Pp ? cp[row * Pp + c] : hx[row * Pp + (c - Pp)];
    pred[i] = f2bf(v);
  }
}

__global__ void k_head_link(const float* __restrict__ t1, const float* __restrict__ W2,
                            const float* __restrict__ b2, float* __restrict__ out) {
  long row = (long)blockIdx.x * blockDim.x + threadIdx.x;
  if (row >= ROWS) return;
  float s = b2[0];
  const float* t = t1 + row * Pp;
  for (int c = 0; c < Pp; ++c) s += t[c] * W2[c];
  out[row * (1 + NC)] = s;
}

__global__ void k_head_label(const float* __restrict__ t2, const float* __restrict__ W2,
                             const float* __restrict__ b2, float* __restrict__ out) {
  long idx = (long)blockIdx.x * blockDim.x + threadIdx.x;
  if (idx >= ROWS * NC) return;
  long row = idx / NC; int cls = (int)(idx % NC);
  float s = b2[cls];
  const float* t = t2 + row * Pp;
  for (int c = 0; c < Pp; ++c) s += t[c] * W2[c * NC + cls];
  out[row * (1 + NC) + 1 + cls] = s;
}

// --------------------------------- host ------------------------------------

extern "C" void kernel_launch(void* const* d_in, const int* in_sizes, int n_in,
                              void* d_out, int out_size, void* d_ws, size_t ws_size,
                              hipStream_t stream) {
  const float* nodes_in = (const float*)d_in[0];
  const float* bias  = (const float*)d_in[1];
  const int*   spk   = (const int*)d_in[2];
  const int*   trn   = (const int*)d_in[3];
  const float* spk_emb = (const float*)d_in[4];
  const float* trn_emb = (const float*)d_in[5];
  const float* pos_emb = (const float*)d_in[6];
  const float* Wq = (const float*)d_in[7],  * bq = (const float*)d_in[8];
  const float* Wk = (const float*)d_in[9],  * bk = (const float*)d_in[10];
  const float* Wv = (const float*)d_in[11], * bv = (const float*)d_in[12];
  const float* Wsk = (const float*)d_in[13], * bsk = (const float*)d_in[14];
  const float* Wsv = (const float*)d_in[15], * bsv = (const float*)d_in[16];
  const float* Wo = (const float*)d_in[17], * bo = (const float*)d_in[18];
  const float* ln_g = (const float*)d_in[19], * ln_b = (const float*)d_in[20];
  const float* pn_g = (const float*)d_in[21], * pn_b = (const float*)d_in[22];
  const float* Wr = (const float*)d_in[23], * br = (const float*)d_in[24];
  const float* Wz = (const float*)d_in[25], * bz = (const float*)d_in[26];
  const float* Wc = (const float*)d_in[27], * bc = (const float*)d_in[28];
  const float* Wu = (const float*)d_in[29], * bu = (const float*)d_in[30];
  const float* lkW1 = (const float*)d_in[31], * lkb1 = (const float*)d_in[32];
  const float* lkW2 = (const float*)d_in[33], * lkb2 = (const float*)d_in[34];
  const float* lbW1 = (const float*)d_in[35], * lbb1 = (const float*)d_in[36];
  const float* lbW2 = (const float*)d_in[37], * lbb2 = (const float*)d_in[38];
  float* out = (float*)d_out;

  size_t off = 0;
  auto alloc = [&](size_t bytes) -> void* {
    off = (off + 255) & ~(size_t)255;
    void* p = (char*)d_ws + off;
    off += bytes;
    return p;
  };
  float* hx      = (float*)alloc(ROWS * Pp * 4);
  float* cpath   = (float*)alloc(ROWS * Pp * 4);
  unsigned short* pn_bf = (unsigned short*)alloc(ROWS * Pp * 2);
  float* skv     = (float*)alloc(ROWS * 128 * 4);            // [sk|sv] interleaved
  unsigned short* pred_bf = (unsigned short*)alloc(ROWS * 512 * 2);
  float* t1      = (float*)alloc(ROWS * Pp * 4);
  float* qb = (float*)alloc(NODES * Hh * 4);
  float* kb_ = (float*)alloc(NODES * Hh * 4);
  float* vb = (float*)alloc(NODES * Hh * 4);
  float* attn_out = (float*)alloc(NODES * Hh * 4);
  float* proj = (float*)alloc(NODES * Hh * 4);
  float* nodes = (float*)alloc(NODES * Hh * 4);
  unsigned short* nodes_bf = (unsigned short*)alloc(NODES * Hh * 2);
  unsigned short* attnout_bf = (unsigned short*)alloc(NODES * Hh * 2);
  unsigned short* WqT = (unsigned short*)alloc((size_t)Hh * Hh * 2);
  unsigned short* WkT = (unsigned short*)alloc((size_t)Hh * Hh * 2);
  unsigned short* WvT = (unsigned short*)alloc((size_t)Hh * Hh * 2);
  unsigned short* WoT = (unsigned short*)alloc((size_t)Hh * Hh * 2);
  unsigned short* WskvT = (unsigned short*)alloc((size_t)128 * Pp * 2); // [Wsk|Wsv]^T
  float* bskv = (float*)alloc(128 * 4);
  unsigned short* WrAT = (unsigned short*)alloc((size_t)Pp * Hh * 2);
  unsigned short* WrBT = (unsigned short*)alloc((size_t)Pp * Hh * 2);
  unsigned short* WzAT = (unsigned short*)alloc((size_t)Pp * Hh * 2);
  unsigned short* WzBT = (unsigned short*)alloc((size_t)Pp * Hh * 2);
  unsigned short* WcAT = (unsigned short*)alloc((size_t)Pp * Hh * 2);
  unsigned short* WcBT = (unsigned short*)alloc((size_t)Pp * Hh * 2);
  unsigned short* WrCT = (unsigned short*)alloc((size_t)Pp * Pp * 2);
  unsigned short* WzCT = (unsigned short*)alloc((size_t)Pp * Pp * 2);
  unsigned short* WuT  = (unsigned short*)alloc((size_t)Pp * Pp * 2);
  unsigned short* lk1T = (unsigned short*)alloc((size_t)Pp * 512 * 2);
  unsigned short* lb1T = (unsigned short*)alloc((size_t)Pp * 512 * 2);
  float* NAr = (float*)alloc(NODES * Pp * 4);
  float* NBr = (float*)alloc(NODES * Pp * 4);
  float* NAz = (float*)alloc(NODES * Pp * 4);
  float* NBz = (float*)alloc(NODES * Pp * 4);
  float* NAc = (float*)alloc(NODES * Pp * 4);
  float* NBc = (float*)alloc(NODES * Pp * 4);

  auto gemm = [&](const unsigned short* A, const unsigned short* Bt, const float* bi,
                  float* C, int M, int K, int Ncols, int act) {
    dim3 g((M + 63) / 64, Ncols / 64);
    k_gemm<<<g, 128, 0, stream>>>(A, Bt, bi, C, M, K, Ncols, act);
  };
  auto prep = [&](const float* W, unsigned short* Bt, int K, int Ncols, int row0, int ldw) {
    k_prep_bt<<<1024, 256, 0, stream>>>(W, Bt, K, Ncols, row0, ldw);
  };

  // ---- one-time prep ----
  k_zero<<<4096, 256, 0, stream>>>(hx, ROWS * Pp);
  k_const_path<<<8192, 256, 0, stream>>>(spk, trn, spk_emb, trn_emb, pos_emb, cpath);
  hipMemcpyAsync(nodes, nodes_in, NODES * Hh * 4, hipMemcpyDeviceToDevice, stream);
  k_cvt_bf<<<1024, 256, 0, stream>>>(nodes_in, nodes_bf, NODES * Hh);
  prep(Wq, WqT, Hh, Hh, 0, Hh);  prep(Wk, WkT, Hh, Hh, 0, Hh);
  prep(Wv, WvT, Hh, Hh, 0, Hh);  prep(Wo, WoT, Hh, Hh, 0, Hh);
  prep(Wsk, WskvT, Pp, DK, 0, DK);
  prep(Wsv, WskvT + (size_t)DK * Pp, Pp, DK, 0, DK);
  hipMemcpyAsync(bskv, bsk, DK * 4, hipMemcpyDeviceToDevice, stream);
  hipMemcpyAsync(bskv + DK, bsv, DK * 4, hipMemcpyDeviceToDevice, stream);
  prep(Wr, WrAT, Hh, Pp, 0, Pp);   prep(Wr, WrBT, Hh, Pp, Hh, Pp);
  prep(Wr, WrCT, Pp, Pp, 2 * Hh, Pp);
  prep(Wz, WzAT, Hh, Pp, 0, Pp);   prep(Wz, WzBT, Hh, Pp, Hh, Pp);
  prep(Wz, WzCT, Pp, Pp, 2 * Hh, Pp);
  prep(Wc, WcAT, Hh, Pp, 0, Pp);   prep(Wc, WcBT, Hh, Pp, Hh, Pp);
  prep(Wu, WuT, Pp, Pp, 0, Pp);
  prep(lkW1, lk1T, 512, Pp, 0, Pp); prep(lbW1, lb1T, 512, Pp, 0, Pp);

  // ---- layers ----
  for (int L = 0; L < LAYERS; ++L) {
    // path LN -> pn (bf16); fused sk|sv GEMM (one pn read)
    k_ln_paths<<<(int)(ROWS / 8), 256, 0, stream>>>(cpath, hx, pn_g, pn_b, pn_bf);
    gemm(pn_bf, WskvT, bskv, skv, (int)ROWS, Pp, 128, 0);
    // qkv
    gemm(nodes_bf, WqT, bq, qb, (int)NODES, Hh, Hh, 0);
    gemm(nodes_bf, WkT, bk, kb_, (int)NODES, Hh, Hh, 0);
    gemm(nodes_bf, WvT, bv, vb, (int)NODES, Hh, Hh, 0);
    // attention
    k_attn<<<(Bq * HEADS * Nn) / 4, 128, 0, stream>>>(qb, kb_, vb, skv, bias, attn_out);
    // out-proj + relu + residual + LN
    k_cvt_bf<<<1024, 256, 0, stream>>>(attn_out, attnout_bf, NODES * Hh);
    gemm(attnout_bf, WoT, bo, proj, (int)NODES, Hh, Hh, 0);
    k_postproj<<<(int)(NODES / 8), 256, 0, stream>>>(proj, nodes, nodes_bf, ln_g, ln_b);
    // node-side GRU projections (pair terms collapse to per-node terms)
    gemm(nodes_bf, WrAT, nullptr, NAr, (int)NODES, Hh, Pp, 0);
    gemm(nodes_bf, WrBT, nullptr, NBr, (int)NODES, Hh, Pp, 0);
    gemm(nodes_bf, WzAT, nullptr, NAz, (int)NODES, Hh, Pp, 0);
    gemm(nodes_bf, WzBT, nullptr, NBz, (int)NODES, Hh, Pp, 0);
    gemm(nodes_bf, WcAT, nullptr, NAc, (int)NODES, Hh, Pp, 0);
    gemm(nodes_bf, WcBT, nullptr, NBc, (int)NODES, Hh, Pp, 0);
    // fused pairwise GRU update (3 WMMA GEMMs + gating, in-place on hx)
    k_path_update<<<(int)(ROWS / 64), 128, 0, stream>>>(
        hx, WrCT, WzCT, WuT, NAr, NBr, NAz, NBz, NAc, NBc, br, bz, bc, bu);
  }

  // ---- heads ----
  k_concat_pred<<<8192, 256, 0, stream>>>(cpath, hx, pred_bf);
  gemm(pred_bf, lk1T, lkb1, t1, (int)ROWS, 512, Pp, 1);         // tanh
  k_head_link<<<(int)((ROWS + 255) / 256), 256, 0, stream>>>(t1, lkW2, lkb2, out);
  gemm(pred_bf, lb1T, lbb1, t1, (int)ROWS, 512, Pp, 1);         // tanh (reuse t1)
  k_head_label<<<(int)((ROWS * NC + 255) / 256), 256, 0, stream>>>(t1, lbW2, lbb2, out);

  (void)in_sizes; (void)n_in; (void)out_size; (void)ws_size;
}
